// SGL_66718021976722
// MI455X (gfx1250) — matrix operand order repaired and checked by reading, not compiled, not run on hardware
//
#include <hip/hip_runtime.h>
#include <hip/hip_bf16.h>

#define U_NUM   100000
#define I_NUM   50000
#define NNODE   (U_NUM + I_NUM)
#define DIM     128
#define BATCH   2048
#define INV_TAU 5.0f            // 1 / 0.2
#define LOG2E   1.44269504088896340736f
#define LMBD_SSL 0.1f
#define LMBD_REG 1e-4f

typedef __attribute__((ext_vector_type(16))) _Float16 v16h;
typedef __attribute__((ext_vector_type(8)))  _Float16 v8h;
typedef __attribute__((ext_vector_type(8)))  float    v8f;

// ---------------------------------------------------------------- utilities
__device__ __forceinline__ float waveRed32(float v) {
    #pragma unroll
    for (int m = 1; m < 32; m <<= 1) v += __shfl_xor(v, m, 32);
    return v;   // butterfly: all lanes hold the total
}

// ------------------------------------------------- acc = layer-0 embeddings
__global__ void k_init_acc(const float* __restrict__ ue, const float* __restrict__ ie,
                           float* __restrict__ acc) {
    size_t base = ((size_t)blockIdx.x * blockDim.x + threadIdx.x) * 4;
    if (base >= (size_t)NNODE * DIM) return;
    size_t node = base / DIM;
    const float* s = (node < U_NUM) ? (ue + base) : (ie + (base - (size_t)U_NUM * DIM));
    *(float4*)(acc + base) = *(const float4*)s;
}

// ------------------------------------------------------- SpMM: dst[r] += v*src[c]
// one edge per 32-lane wave (8 per 256-thread block), 4 dims per lane (f32 atomics, L2-resident)
__global__ void k_spmm(const int* __restrict__ row, const int* __restrict__ col,
                       const float* __restrict__ val, int E,
                       const float* __restrict__ src,
                       const float* __restrict__ ue, const float* __restrict__ ie,
                       int layer0, float* __restrict__ dst) {
    int e = blockIdx.x * 8 + (threadIdx.x >> 5);
    if (e >= E) return;
    int lane = threadIdx.x & 31;
    int r = row[e], c = col[e];
    float v = val[e];
    const float* s;
    if (layer0) s = (c < U_NUM) ? (ue + (size_t)c * DIM) : (ie + (size_t)(c - U_NUM) * DIM);
    else        s = src + (size_t)c * DIM;
    float4 x = *(const float4*)(s + lane * 4);
    float* d = dst + (size_t)r * DIM + lane * 4;
    atomicAdd(d + 0, v * x.x);
    atomicAdd(d + 1, v * x.y);
    atomicAdd(d + 2, v * x.z);
    atomicAdd(d + 3, v * x.w);
}

// --------------------------------------------------------------- acc += x
__global__ void k_add(float* __restrict__ acc, const float* __restrict__ x) {
    size_t i = ((size_t)blockIdx.x * blockDim.x + threadIdx.x) * 4;
    if (i >= (size_t)NNODE * DIM) return;
    float4 a = *(float4*)(acc + i);
    float4 b = *(const float4*)(x + i);
    a.x += b.x; a.y += b.y; a.z += b.z; a.w += b.w;
    *(float4*)(acc + i) = a;
}

// ------------------------------------------- BPR + L2-reg partial sums (1 wave / sample)
__global__ void k_bpr(const float* __restrict__ acc,
                      const float* __restrict__ ue, const float* __restrict__ ie,
                      const int* __restrict__ uid, const int* __restrict__ iid,
                      const int* __restrict__ nid,
                      float* bpr_acc, float* reg_acc) {
    int b = blockIdx.x * (blockDim.x >> 5) + (threadIdx.x >> 5);
    if (b >= BATCH) return;
    int lane = threadIdx.x & 31;
    int u = uid[b], ip = iid[b], nn = nid[b];
    const float* fu = acc + (size_t)u * DIM + lane * 4;
    const float* fp = acc + (size_t)(U_NUM + ip) * DIM + lane * 4;
    const float* fn = acc + (size_t)(U_NUM + nn) * DIM + lane * 4;
    float4 a = *(const float4*)fu, p = *(const float4*)fp, q = *(const float4*)fn;
    float pos = a.x*p.x + a.y*p.y + a.z*p.z + a.w*p.w;
    float neg = a.x*q.x + a.y*q.y + a.z*q.z + a.w*q.w;
    const float* e0 = ue + (size_t)u  * DIM + lane * 4;
    const float* e1 = ie + (size_t)ip * DIM + lane * 4;
    const float* e2 = ie + (size_t)nn * DIM + lane * 4;
    float4 x0 = *(const float4*)e0, x1 = *(const float4*)e1, x2 = *(const float4*)e2;
    float reg = x0.x*x0.x + x0.y*x0.y + x0.z*x0.z + x0.w*x0.w
              + x1.x*x1.x + x1.y*x1.y + x1.z*x1.z + x1.w*x1.w
              + x2.x*x2.x + x2.y*x2.y + x2.z*x2.z + x2.w*x2.w;
    pos = waveRed32(pos); neg = waveRed32(neg); reg = waveRed32(reg);
    if (lane == 0) {
        float x = (pos - neg) * 0.0625f;              // acc is 4x mean -> dot is 16x
        float loss = fmaxf(-x, 0.f) + log1pf(__expf(-fabsf(x)));  // -log_sigmoid(x)
        atomicAdd(bpr_acc, loss);
        atomicAdd(reg_acc, reg);
    }
}

// ------------------- view-1: normalize gathered batch rows -> f32 (for pos) + f16 (WMMA A)
__global__ void k_norm_gather(const float* __restrict__ acc,
                              const int* __restrict__ uid, const int* __restrict__ iid,
                              float* __restrict__ ub1f, float* __restrict__ ib1f,
                              _Float16* __restrict__ ub1h, _Float16* __restrict__ ib1h) {
    int w = blockIdx.x * (blockDim.x >> 5) + (threadIdx.x >> 5);
    if (w >= 2 * BATCH) return;
    int lane = threadIdx.x & 31;
    int isItem = (w >= BATCH);
    int b = isItem ? (w - BATCH) : w;
    size_t node = isItem ? ((size_t)U_NUM + iid[b]) : (size_t)uid[b];
    float4 x = *(const float4*)(acc + node * DIM + lane * 4);
    float ss = waveRed32(x.x*x.x + x.y*x.y + x.z*x.z + x.w*x.w);
    float rn = rsqrtf(fmaxf(ss, 1e-24f));
    float*    df = (isItem ? ib1f : ub1f) + (size_t)b * DIM + lane * 4;
    _Float16* dh = (isItem ? ib1h : ub1h) + (size_t)b * DIM + lane * 4;
    float y0 = x.x*rn, y1 = x.y*rn, y2 = x.z*rn, y3 = x.w*rn;
    df[0]=y0; df[1]=y1; df[2]=y2; df[3]=y3;
    dh[0]=(_Float16)y0; dh[1]=(_Float16)y1; dh[2]=(_Float16)y2; dh[3]=(_Float16)y3;
}

// ------------------- view-2: normalize ALL rows -> f16 (WMMA B); scale-invariant so no /4
__global__ void k_norm_all(const float* __restrict__ acc,
                           _Float16* __restrict__ u2h, _Float16* __restrict__ i2h) {
    int n = blockIdx.x * (blockDim.x >> 5) + (threadIdx.x >> 5);
    if (n >= NNODE) return;
    int lane = threadIdx.x & 31;
    float4 x = *(const float4*)(acc + (size_t)n * DIM + lane * 4);
    float ss = waveRed32(x.x*x.x + x.y*x.y + x.z*x.z + x.w*x.w);
    float rn = rsqrtf(fmaxf(ss, 1e-24f));
    _Float16* d = ((n < U_NUM) ? (u2h + (size_t)n * DIM)
                               : (i2h + (size_t)(n - U_NUM) * DIM)) + lane * 4;
    d[0]=(_Float16)(x.x*rn); d[1]=(_Float16)(x.y*rn);
    d[2]=(_Float16)(x.z*rn); d[3]=(_Float16)(x.w*rn);
}

// ------------------------------------------------ pos_b = <view1_b , view2[id_b]>
__global__ void k_pos(const float* __restrict__ ub1f, const float* __restrict__ ib1f,
                      const int* __restrict__ uid, const int* __restrict__ iid,
                      const _Float16* __restrict__ u2h, const _Float16* __restrict__ i2h,
                      float* __restrict__ pos_u, float* __restrict__ pos_i) {
    int w = blockIdx.x * (blockDim.x >> 5) + (threadIdx.x >> 5);
    if (w >= 2 * BATCH) return;
    int lane = threadIdx.x & 31;
    int isItem = (w >= BATCH);
    int b = isItem ? (w - BATCH) : w;
    const float* a = (isItem ? ib1f : ub1f) + (size_t)b * DIM + lane * 4;
    const _Float16* h = (isItem ? (i2h + (size_t)iid[b] * DIM)
                                : (u2h + (size_t)uid[b] * DIM)) + lane * 4;
    float d = a[0]*(float)h[0] + a[1]*(float)h[1] + a[2]*(float)h[2] + a[3]*(float)h[3];
    d = waveRed32(d);
    if (lane == 0) (isItem ? pos_i : pos_u)[b] = d;
}

// ---------------- SSL GEMM + streaming exp-sum:  S[m] += sum_n exp((A·B^T - pos_m)/tau)
// A: [BATCH x 128] f16 (view-1 batch rows), B: [ncols x 128] f16 (all view-2 rows).
// M-blocked x2: each block covers 32 batch rows, so each loaded B fragment feeds 2 WMMAs.
// exp((c-p)/tau) folded to exp2(fma(c, K, -p*K)) -> 1 FMA + 1 TRANS per element.
__global__ __launch_bounds__(128)
void k_ssl(const _Float16* __restrict__ Ah, const _Float16* __restrict__ Bh,
           const float* __restrict__ pos, int ncols, float* __restrict__ S) {
    const int lane = threadIdx.x & 31;
    // wave index is uniform across the wave: tell the compiler -> scalar tile loop
    const int wave = __builtin_amdgcn_readfirstlane(threadIdx.x >> 5);
    const int half = lane >> 4;        // 0: lanes 0-15, 1: lanes 16-31
    const int nn   = lane & 15;
    const int rowBase = blockIdx.x * 32;
    const float K2 = INV_TAU * LOG2E;

    // A fragments: 16-bit A 16x32 layout -- lane holds row M=lane%16,
    // halves 0..15 = K(half*16 + 0..15) (contiguous, matches B)
    v16h a[2][4];
    #pragma unroll
    for (int t = 0; t < 2; ++t) {
        const _Float16* arow = Ah + (size_t)(rowBase + 16*t + nn) * DIM;
        #pragma unroll
        for (int f = 0; f < 4; ++f) {
            a[t][f] = *(const v16h*)(arow + 32*f + 16*half);
        }
    }
    // pre-scaled -pos*K for this half-wave's output rows (M = v + 8*half) per M-tile
    float ps2[2][8], acc8[2][8];
    #pragma unroll
    for (int t = 0; t < 2; ++t)
        #pragma unroll
        for (int v = 0; v < 8; ++v) {
            ps2[t][v] = -pos[rowBase + 16*t + v + 8*half] * K2;
            acc8[t][v] = 0.f;
        }

    const int nTiles = ncols >> 4;
    const int stride = (int)gridDim.y * 4;
    for (int ct = blockIdx.y * 4 + wave; ct < nTiles; ct += stride) {
        // B 32x16 layout: lane holds col N=lane%16; halves = 16 consecutive K per half-wave.
        const _Float16* brow = Bh + (size_t)(ct * 16 + nn) * DIM + 16 * half;
        if (ct + stride < nTiles)
            __builtin_prefetch(Bh + (size_t)((ct + stride) * 16 + nn) * DIM, 0, 0);
        v16h b[4];
        #pragma unroll
        for (int f = 0; f < 4; ++f) b[f] = *(const v16h*)(brow + 32 * f);

        v8f c0 = {}, c1 = {};
        #pragma unroll
        for (int f = 0; f < 4; ++f) {
            c0 = __builtin_amdgcn_wmma_f32_16x16x32_f16(false, a[0][f], false, b[f],
                                                        (short)0, c0, false, false);
            c1 = __builtin_amdgcn_wmma_f32_16x16x32_f16(false, a[1][f], false, b[f],
                                                        (short)0, c1, false, false);
        }
        // D layout: VGPR v, lane -> (M = v + 8*half, N = lane%16). Bounded logits -> no max.
        #pragma unroll
        for (int v = 0; v < 8; ++v) {
            acc8[0][v] += exp2f(fmaf(c0[v], K2, ps2[0][v]));
            acc8[1][v] += exp2f(fmaf(c1[v], K2, ps2[1][v]));
        }
    }
    // reduce over the 16 N-lanes of this half-wave, then one atomic per output row
    #pragma unroll
    for (int t = 0; t < 2; ++t)
        #pragma unroll
        for (int v = 0; v < 8; ++v) {
            float s = acc8[t][v];
            #pragma unroll
            for (int m = 1; m < 16; m <<= 1) s += __shfl_xor(s, m, 32);
            if (nn == 0) atomicAdd(&S[rowBase + 16*t + v + 8*half], s);
        }
}

// ------------------------------------------------------------- final combine
__global__ void k_final(const float* __restrict__ Su, const float* __restrict__ Si,
                        const float* __restrict__ bpr_acc, const float* __restrict__ reg_acc,
                        float* __restrict__ out) {
    __shared__ float red[256];
    float s = 0.f;
    for (int b = threadIdx.x; b < BATCH; b += 256)
        s += __logf(Su[b]) + __logf(Si[b]);
    red[threadIdx.x] = s; __syncthreads();
    for (int off = 128; off; off >>= 1) {
        if ((int)threadIdx.x < off) red[threadIdx.x] += red[threadIdx.x + off];
        __syncthreads();
    }
    if (threadIdx.x == 0)
        out[0] = bpr_acc[0] * (1.f / BATCH) + LMBD_SSL * red[0]
               + LMBD_REG * (reg_acc[0] * (1.f / (2.f * BATCH)));
}

// ============================================================== host driver
extern "C" void kernel_launch(void* const* d_in, const int* in_sizes, int n_in,
                              void* d_out, int out_size, void* d_ws, size_t ws_size,
                              hipStream_t stream) {
    (void)n_in; (void)out_size; (void)ws_size;
    const float* ue    = (const float*)d_in[0];
    const float* ie    = (const float*)d_in[1];
    const int*   gr[3] = {(const int*)d_in[2], (const int*)d_in[5], (const int*)d_in[8]};
    const int*   gc[3] = {(const int*)d_in[3], (const int*)d_in[6], (const int*)d_in[9]};
    const float* gv[3] = {(const float*)d_in[4], (const float*)d_in[7], (const float*)d_in[10]};
    const int    gE[3] = {in_sizes[2], in_sizes[5], in_sizes[8]};
    const int* uid = (const int*)d_in[11];
    const int* iid = (const int*)d_in[12];
    const int* nid = (const int*)d_in[13];

    const size_t ND = (size_t)NNODE * DIM;
    float* bufA = (float*)d_ws;
    float* bufB = bufA + ND;
    float* acc  = bufB + ND;
    float* ub1f = acc + ND;
    float* ib1f = ub1f + (size_t)BATCH * DIM;
    float* pos_u = ib1f + (size_t)BATCH * DIM;
    float* pos_i = pos_u + BATCH;
    float* Su = pos_i + BATCH;
    float* Si = Su + BATCH;
    float* bpr_acc = Si + BATCH;
    float* reg_acc = bpr_acc + 1;
    size_t foff = (size_t)((reg_acc + 1) - (float*)d_ws);
    foff = (foff + 15) & ~(size_t)15;              // 64-byte align the f16 region
    _Float16* ub1h = (_Float16*)((float*)d_ws + foff);
    _Float16* ib1h = ub1h + (size_t)BATCH * DIM;
    _Float16* u2h  = ib1h + (size_t)BATCH * DIM;
    _Float16* i2h  = u2h + (size_t)U_NUM * DIM;

    const int elemBlocks = (int)((ND / 4 + 255) / 256);

    auto propagate = [&](int g) {
        const int *r = gr[g], *c = gc[g]; const float* v = gv[g]; int E = gE[g];
        k_init_acc<<<elemBlocks, 256, 0, stream>>>(ue, ie, acc);
        hipMemsetAsync(bufA, 0, ND * sizeof(float), stream);
        k_spmm<<<(E + 7) / 8, 256, 0, stream>>>(r, c, v, E, nullptr, ue, ie, 1, bufA);
        k_add<<<elemBlocks, 256, 0, stream>>>(acc, bufA);
        hipMemsetAsync(bufB, 0, ND * sizeof(float), stream);
        k_spmm<<<(E + 7) / 8, 256, 0, stream>>>(r, c, v, E, bufA, ue, ie, 0, bufB);
        k_add<<<elemBlocks, 256, 0, stream>>>(acc, bufB);
        hipMemsetAsync(bufA, 0, ND * sizeof(float), stream);
        k_spmm<<<(E + 7) / 8, 256, 0, stream>>>(r, c, v, E, bufB, ue, ie, 0, bufA);
        k_add<<<elemBlocks, 256, 0, stream>>>(acc, bufA);
    };

    // zero Su, Si, bpr_acc, reg_acc (contiguous)
    hipMemsetAsync(Su, 0, (2 * BATCH + 2) * sizeof(float), stream);

    // full graph -> BPR + reg
    propagate(0);
    k_bpr<<<BATCH / 4, 128, 0, stream>>>(acc, ue, ie, uid, iid, nid, bpr_acc, reg_acc);

    // view 1 -> normalized batch rows
    propagate(1);
    k_norm_gather<<<(2 * BATCH) / 4, 128, 0, stream>>>(acc, uid, iid, ub1f, ib1f, ub1h, ib1h);

    // view 2 -> normalized full tables + positives
    propagate(2);
    k_norm_all<<<(NNODE + 3) / 4, 128, 0, stream>>>(acc, u2h, i2h);
    k_pos<<<(2 * BATCH) / 4, 128, 0, stream>>>(ub1f, ib1f, uid, iid, u2h, i2h, pos_u, pos_i);

    // SSL InfoNCE via WMMA streaming exp-sum (32 batch rows per block)
    dim3 blk(128);
    k_ssl<<<dim3(BATCH / 32, 64), blk, 0, stream>>>(ub1h, u2h, pos_u, U_NUM, Su);
    k_ssl<<<dim3(BATCH / 32, 64), blk, 0, stream>>>(ib1h, i2h, pos_i, I_NUM, Si);

    k_final<<<1, 256, 0, stream>>>(Su, Si, bpr_acc, reg_acc, (float*)d_out);
}
